// WindowAttention_55748675502914
// MI455X (gfx1250) — matrix operand order repaired and checked by reading, not compiled, not run on hardware
//
#include <hip/hip_runtime.h>

#define WH 8
#define WW 8
#define N_TOK 64
#define DIM 128
#define NUM_HEADS 4
#define HEAD_DIM 32
#define QK_SCALE 0.17677669529663687f   // 32^-0.5

typedef __bf16 bf16;
typedef __attribute__((ext_vector_type(16))) __bf16 v16bf;
typedef __attribute__((ext_vector_type(4)))  __bf16 v4bf;
typedef __attribute__((ext_vector_type(8)))  float  v8f;
typedef __attribute__((ext_vector_type(4)))  float  v4f;

union FragBF {
    v16bf v;
    v4f   q[2];
};

__device__ __forceinline__ int rfl(int x) { return __builtin_amdgcn_readfirstlane(x); }

// A-matrix 16x32 bf16 fragment from row-major source (element (m,k) = src[m*stride + k]).
// ISA layout: lanes 0-15 row m, K in {hi*8..hi*8+7} (VGPR0-3) and {16+hi*8..} (VGPR4-7).
__device__ __forceinline__ v16bf load_frag_a(const bf16* src, int stride, int lane) {
    const int row = lane & 15;
    const int hi  = (lane >> 4) & 1;
    const bf16* p = src + row * stride + hi * 8;
    FragBF f;
    f.q[0] = *(const v4f*)(p);
    f.q[1] = *(const v4f*)(p + 16);
    return f.v;
}

// B-matrix 32x16 bf16 fragment where B[k,n] = w[n*stride + k] (i.e. B = Wᵀ, w row-major).
// ISA layout: lane n holds column n; K in {kh*8..kh*8+7} (VGPR0-3) and {16+kh*8..} (VGPR4-7).
__device__ __forceinline__ v16bf load_frag_bT(const bf16* w, int stride, int lane) {
    const int col = lane & 15;
    const int kh  = (lane >> 4) & 1;
    const bf16* p = w + col * stride + kh * 8;
    FragBF f;
    f.q[0] = *(const v4f*)(p);
    f.q[1] = *(const v4f*)(p + 16);
    return f.v;
}

#define WMMA_BF16(a, b, c) \
    __builtin_amdgcn_wmma_f32_16x16x32_bf16(false, (a), false, (b), (short)0, (c), false, false)

// ---------------- prep: weights -> bf16, dense relative-position bias ----------------
__global__ void winattn_prep_kernel(const float* __restrict__ qkv_w,
                                    const float* __restrict__ proj_w,
                                    const float* __restrict__ bias_table,
                                    bf16* __restrict__ wq,
                                    bf16* __restrict__ wp,
                                    float* __restrict__ bias_full) {
    int t = blockIdx.x * blockDim.x + threadIdx.x;
    const int NQ = 3 * DIM * DIM;             // 49152
    const int NP = DIM * DIM;                 // 16384
    const int NB = NUM_HEADS * N_TOK * N_TOK; // 16384
    if (t < NQ) {
        wq[t] = (bf16)qkv_w[t];
    } else if (t < NQ + NP) {
        int i = t - NQ;
        wp[i] = (bf16)proj_w[i];
    } else if (t < NQ + NP + NB) {
        int i  = t - NQ - NP;
        int h  = i >> 12;
        int ii = (i >> 6) & 63;
        int jj = i & 63;
        int yi = ii >> 3, xi = ii & 7, yj = jj >> 3, xj = jj & 7;
        int ridx = (yi - yj + WH - 1) * (2 * WW - 1) + (xi - xj + WW - 1);
        bias_full[i] = bias_table[ridx * NUM_HEADS + h];
    }
}

// ---------------- fused window attention, one block per window ----------------
// LDS plan (bytes):
//   [0,16K)      Xs   bf16 64x128         (dead after phase 1)
//   [16K,32K)    Qs   bf16 64x128 (scaled)
//   [32K,48K)    Ks   bf16 64x128
//   [48K,64K)    Vt   bf16 128x64 (V transposed)
//   [64K,132K)   At   f32  4x64x68 (padded, 16B-aligned rows) -> overlaid by Os in phase 4
//   [0,36K)      Ab   bf16 4x64x72 (padded) overlays dead Xs/Qs/Ks
#define OFF_X   0
#define OFF_Q   16384
#define OFF_K   32768
#define OFF_V   49152
#define OFF_AT  65536
#define OFF_AB  0
#define OFF_O   65536
#define ATS     68
#define ABS     72
#define SMEM_BYTES (65536 + NUM_HEADS * N_TOK * ATS * 4)   // 135168

__launch_bounds__(256)
__global__ void winattn_main_kernel(const float* __restrict__ x,
                                    const float* __restrict__ mask,
                                    const float* __restrict__ qkv_b,
                                    const float* __restrict__ proj_b,
                                    const bf16* __restrict__ wq,
                                    const bf16* __restrict__ wp,
                                    const float* __restrict__ bias_full,
                                    float* __restrict__ out,
                                    int nW) {
    extern __shared__ unsigned char smem[];
    bf16*  Xs = (bf16*)(smem + OFF_X);
    bf16*  Qs = (bf16*)(smem + OFF_Q);
    bf16*  Ks = (bf16*)(smem + OFF_K);
    bf16*  Vt = (bf16*)(smem + OFF_V);
    float* At = (float*)(smem + OFF_AT);
    bf16*  Ab = (bf16*)(smem + OFF_AB);
    bf16*  Os = (bf16*)(smem + OFF_O);

    const int b    = blockIdx.x;
    const int w    = b % nW;
    const int tid  = threadIdx.x;
    const int lane = tid & 31;
    const int wave = rfl(tid >> 5);        // wave-uniform -> SGPR
    const int c16  = lane & 15;
    const int hi   = (lane >> 4) & 1;

    // ---- phase 0: stage X as bf16 (vectorized: float4 in, packed b64 out) ----
    {
        const v4f* xg4 = (const v4f*)(x + (size_t)b * (N_TOK * DIM));
        for (int t = tid; t < (N_TOK * DIM) / 4; t += 256) {
            v4f xv = xg4[t];
            v4bf pk;
            pk[0] = (bf16)xv[0]; pk[1] = (bf16)xv[1];
            pk[2] = (bf16)xv[2]; pk[3] = (bf16)xv[3];
            *(v4bf*)(Xs + 4 * t) = pk;
        }
    }
    __syncthreads();

    // ---- phase 1: QKV = X @ Wqkvᵀ + b   (M=64, N=384, K=128) ----
    for (int t = 0; t < 12; ++t) {
        const int idx = rfl(wave + 8 * t);     // 0..95 tiles, SGPR
        const int nt = idx % 24, mt = idx / 24;
        v8f acc = {};
#pragma unroll
        for (int kk = 0; kk < 4; ++kk) {
            v16bf a  = load_frag_a(Xs + mt * 16 * DIM + kk * 32, DIM, lane);
            v16bf bm = load_frag_bT(wq + (nt * 16) * DIM + kk * 32, DIM, lane);
            acc = WMMA_BF16(a, bm, acc);
        }
        const int c = nt * 16 + c16;           // output channel 0..383
        const float bias = qkv_b[c];
        if (nt < 8) {                          // scalar branch: Q block
#pragma unroll
            for (int v = 0; v < 8; ++v) {
                int tok = mt * 16 + v + 8 * hi;
                Qs[tok * DIM + c] = (bf16)((acc[v] + bias) * QK_SCALE);
            }
        } else if (nt < 16) {                  // K block
            int cc = c - DIM;
#pragma unroll
            for (int v = 0; v < 8; ++v) {
                int tok = mt * 16 + v + 8 * hi;
                Ks[tok * DIM + cc] = (bf16)(acc[v] + bias);
            }
        } else {                               // V block (store transposed)
            int cc = c - 2 * DIM;
#pragma unroll
            for (int v = 0; v < 8; ++v) {
                int tok = mt * 16 + v + 8 * hi;
                Vt[cc * N_TOK + tok] = (bf16)(acc[v] + bias);
            }
        }
    }
    __syncthreads();

    // ---- phase 2: attn = Q @ Kᵀ + rel_bias + mask  (per head M=64,N=64,K=32) ----
    const float* maskw = mask + (size_t)w * N_TOK * N_TOK;
    for (int t = 0; t < 8; ++t) {
        const int idx = rfl(wave + 8 * t);     // 0..63 tiles, SGPR
        const int h = idx >> 4, rem = idx & 15;
        const int mt = rem >> 2, nt = rem & 3;
        v16bf a  = load_frag_a(Qs + mt * 16 * DIM + h * HEAD_DIM, DIM, lane);
        v16bf bm = load_frag_bT(Ks + nt * 16 * DIM + h * HEAD_DIM, DIM, lane);
        v8f acc = {};
        acc = WMMA_BF16(a, bm, acc);
        const int j = nt * 16 + c16;
        const float* bh = bias_full + h * N_TOK * N_TOK;
        float* ah = At + h * N_TOK * ATS;
#pragma unroll
        for (int v = 0; v < 8; ++v) {
            int i = mt * 16 + v + 8 * hi;
            ah[i * ATS + j] = acc[v] + bh[i * N_TOK + j] + maskw[i * N_TOK + j];
        }
    }
    __syncthreads();

    // ---- phase 3: row softmax (256 rows / 256 threads), row held in registers ----
    {
        const int h = tid >> 6, i = tid & 63;
        const float* ah = At + (h * N_TOK + i) * ATS;
        v4f e[16];
#pragma unroll
        for (int j = 0; j < 16; ++j) e[j] = *(const v4f*)(ah + 4 * j);
        float m = -3.0e38f;
#pragma unroll
        for (int j = 0; j < 16; ++j)
            m = fmaxf(m, fmaxf(fmaxf(e[j][0], e[j][1]), fmaxf(e[j][2], e[j][3])));
        float s = 0.f;
#pragma unroll
        for (int j = 0; j < 16; ++j) {
#pragma unroll
            for (int k = 0; k < 4; ++k) {
                float ev = __expf(e[j][k] - m);
                e[j][k] = ev;
                s += ev;
            }
        }
        const float r = 1.0f / s;
        bf16* abr = Ab + (h * N_TOK + i) * ABS;
#pragma unroll
        for (int j = 0; j < 16; ++j) {
            v4bf pk;
            pk[0] = (bf16)(e[j][0] * r); pk[1] = (bf16)(e[j][1] * r);
            pk[2] = (bf16)(e[j][2] * r); pk[3] = (bf16)(e[j][3] * r);
            *(v4bf*)(abr + 4 * j) = pk;
        }
    }
    __syncthreads();

    // ---- phase 4: O = P @ V   (per head M=64,N=32,K=64) ----
    for (int t = 0; t < 4; ++t) {
        const int idx = rfl(wave + 8 * t);     // 0..31 tiles, SGPR
        const int h = idx >> 3, rem = idx & 7;
        const int mt = rem >> 1, nt = rem & 1;
        v8f acc = {};
#pragma unroll
        for (int kt = 0; kt < 2; ++kt) {
            v16bf a  = load_frag_a(Ab + (h * N_TOK + mt * 16) * ABS + kt * 32, ABS, lane);
            v16bf bm = load_frag_bT(Vt + (h * HEAD_DIM + nt * 16) * N_TOK + kt * 32, N_TOK, lane);
            acc = WMMA_BF16(a, bm, acc);
        }
        const int d = nt * 16 + c16;
#pragma unroll
        for (int v = 0; v < 8; ++v) {
            int i = mt * 16 + v + 8 * hi;
            Os[i * DIM + h * HEAD_DIM + d] = (bf16)acc[v];
        }
    }
    __syncthreads();

    // ---- phase 5: out = O @ Wpᵀ + proj_b   (M=64,N=128,K=128) ----
    float* og = out + (size_t)b * (N_TOK * DIM);
    for (int t = 0; t < 4; ++t) {
        const int idx = rfl(wave + 8 * t);     // 0..31 tiles, SGPR
        const int nt = idx & 7, mt = idx >> 3;
        v8f acc = {};
#pragma unroll
        for (int kk = 0; kk < 4; ++kk) {
            v16bf a  = load_frag_a(Os + mt * 16 * DIM + kk * 32, DIM, lane);
            v16bf bm = load_frag_bT(wp + (nt * 16) * DIM + kk * 32, DIM, lane);
            acc = WMMA_BF16(a, bm, acc);
        }
        const int c = nt * 16 + c16;
        const float pb = proj_b[c];
#pragma unroll
        for (int v = 0; v < 8; ++v) {
            int i = mt * 16 + v + 8 * hi;
            og[i * DIM + c] = acc[v] + pb;
        }
    }
}

// ---------------- launcher ----------------
extern "C" void kernel_launch(void* const* d_in, const int* in_sizes, int n_in,
                              void* d_out, int out_size, void* d_ws, size_t ws_size,
                              hipStream_t stream) {
    const float* x      = (const float*)d_in[0];
    const float* mask   = (const float*)d_in[1];
    const float* qkv_w  = (const float*)d_in[2];
    const float* qkv_b  = (const float*)d_in[3];
    const float* proj_w = (const float*)d_in[4];
    const float* proj_b = (const float*)d_in[5];
    const float* bias_t = (const float*)d_in[6];
    float* out = (float*)d_out;

    // workspace: bf16 weights + dense bias table (L2-resident, shared by all blocks)
    bf16*  wq        = (bf16*)d_ws;                            // 98304 B
    bf16*  wp        = (bf16*)((char*)d_ws + 98304);           // 32768 B
    float* bias_full = (float*)((char*)d_ws + 131072);         // 65536 B

    const int B_ = in_sizes[0] / (N_TOK * DIM);                // 4096
    const int nW = in_sizes[1] / (N_TOK * N_TOK);              // 64

    const int prep_elems = 3 * DIM * DIM + DIM * DIM + NUM_HEADS * N_TOK * N_TOK;
    winattn_prep_kernel<<<(prep_elems + 255) / 256, 256, 0, stream>>>(
        qkv_w, proj_w, bias_t, wq, wp, bias_full);

    hipFuncSetAttribute((const void*)winattn_main_kernel,
                        hipFuncAttributeMaxDynamicSharedMemorySize, SMEM_BYTES);
    winattn_main_kernel<<<B_, 256, SMEM_BYTES, stream>>>(
        x, mask, qkv_b, proj_b, wq, wp, bias_full, out, nW);
}
